// GOFADecoderLayer_9216999817473
// MI455X (gfx1250) — matrix operand order repaired and checked by reading, not compiled, not run on hardware
//
#include <hip/hip_runtime.h>
#include <hip/hip_bf16.h>
#include <cstdint>
#include <cstddef>

// ---------------- problem constants ----------------
#define NNODES 20000
#define NEDGES 40000
#define QL     4
#define HID    512
#define NH     8
#define KVH    2
#define HD     64
#define INTERD 2048
#define NROWS  (NNODES * QL)     // 80000
#define EROWS  (NEDGES * QL)     // 160000
#define KVW    (KVH * HD)        // 128

typedef __attribute__((ext_vector_type(16))) __bf16 v16bf;
typedef __attribute__((ext_vector_type(8)))  __bf16 v8bf;
typedef __attribute__((ext_vector_type(8)))  float  v8f;

// fp32 -> bf16 round-to-nearest-even (bits)
__device__ __forceinline__ unsigned short f2bf(float f) {
    unsigned int u = __float_as_uint(f);
    unsigned int r = (u + 0x7FFFu + ((u >> 16) & 1u)) >> 16;
    return (unsigned short)r;
}

// fast hardware transcendental paths
__device__ __forceinline__ float fast_rcp(float x)  { return __builtin_amdgcn_rcpf(x); }
__device__ __forceinline__ float fast_exp2(float x) { return __builtin_amdgcn_exp2f(x); }
__device__ __forceinline__ float fast_exp(float x)  { return __builtin_amdgcn_exp2f(x * 1.44269504089f); }

// ---------------- WMMA fragment loads (ISA 7.12.2 layouts) ----------------
// A 16x32 bf16: lane m=lane&15, half=lane>>4; VGPR0..3 = K {half*8 .. +7},
// VGPR4..7 = K {16+half*8 .. +7}  -> two contiguous 8-element (16B) runs.
__device__ __forceinline__ v16bf load_a_frag(const unsigned short* __restrict__ A,
                                             int K, int row, int half, int k0) {
    const unsigned short* p = A + (size_t)row * K + k0 + half * 8;
    union { v16bf v; v8bf h[2]; } u;
    u.h[0] = *(const v8bf*)(const void*)(p);
    u.h[1] = *(const v8bf*)(const void*)(p + 16);
    return u.v;
}
// B 32x16 bf16 (B[k][n] = W[n][k]): lane n=lane&15, half=lane>>4;
// K {half*8 .. +7} and K {16+half*8 .. +7} from W row (n0+n) (two 16B loads).
__device__ __forceinline__ v16bf load_b_frag(const unsigned short* __restrict__ W,
                                             int K, int n0, int lane, int k0) {
    const unsigned short* p = W + (size_t)(n0 + (lane & 15)) * K + k0 + ((lane >> 4) << 3);
    union { v16bf v; v8bf h[2]; } u;
    u.h[0] = *(const v8bf*)(const void*)(p);
    u.h[1] = *(const v8bf*)(const void*)(p + 16);
    return u.v;
}

// ---------------- GEMM: out[M,N] = A[M,K](bf16) @ W[N,K]^T(bf16) (+resid) ---
// block = 256 thr = 8 waves (2Mx4N), wave tile 64x32 (4x2 WMMA), block 128x128.
template<bool RESID>
__global__ __launch_bounds__(256) void k_gemm(const unsigned short* __restrict__ A,
                                              const unsigned short* __restrict__ W,
                                              const float* __restrict__ resid,
                                              float* __restrict__ out,
                                              int M, int N, int K) {
    const int lane  = threadIdx.x & 31;
    const int wave  = threadIdx.x >> 5;
    const int m0    = blockIdx.x * 128 + (wave & 1) * 64;
    const int n0    = blockIdx.y * 128 + (wave >> 1) * 32;
    const int mlane = lane & 15, half = lane >> 4;

    v8f acc[4][2];
#pragma unroll
    for (int i = 0; i < 4; ++i)
#pragma unroll
        for (int j = 0; j < 2; ++j) acc[i][j] = (v8f)(0.0f);

    for (int k0 = 0; k0 < K; k0 += 32) {
        if (k0 + 32 < K)  // speculative prefetch of next A slab -> global_prefetch_b8
            __builtin_prefetch(A + (size_t)(m0 + mlane) * K + k0 + 32, 0, 0);
        v16bf a[4], b[2];
#pragma unroll
        for (int i = 0; i < 4; ++i) a[i] = load_a_frag(A, K, m0 + i * 16 + mlane, half, k0);
#pragma unroll
        for (int j = 0; j < 2; ++j) b[j] = load_b_frag(W, K, n0 + j * 16, lane, k0);
#pragma unroll
        for (int i = 0; i < 4; ++i)
#pragma unroll
            for (int j = 0; j < 2; ++j)
                acc[i][j] = __builtin_amdgcn_wmma_f32_16x16x32_bf16(
                    false, a[i], false, b[j], (short)0, acc[i][j], false, false);
    }
    // D layout: lane = (m&8)?16+n:n ; VGPR r -> m = half*8 + r
#pragma unroll
    for (int i = 0; i < 4; ++i)
#pragma unroll
        for (int j = 0; j < 2; ++j)
#pragma unroll
            for (int r = 0; r < 8; ++r) {
                int m = m0 + i * 16 + half * 8 + r;
                int n = n0 + j * 16 + mlane;
                size_t idx = (size_t)m * N + n;
                float v = acc[i][j][r];
                if (RESID) v += resid[idx];
                out[idx] = v;
            }
}

// ---------------- fused gate/up GEMM + SiLU, bf16 out ----------------------
// 8 waves (4Mx2N), wave tile 32x32 (2x2 WMMA, two acc sets), block 128x64.
__global__ __launch_bounds__(256) void k_gemm_gateup(const unsigned short* __restrict__ A,
                                                     const unsigned short* __restrict__ Wg,
                                                     const unsigned short* __restrict__ Wu,
                                                     unsigned short* __restrict__ out,
                                                     int M, int N, int K) {
    const int lane  = threadIdx.x & 31;
    const int wave  = threadIdx.x >> 5;
    const int m0    = blockIdx.x * 128 + (wave & 3) * 32;
    const int n0    = blockIdx.y * 64 + (wave >> 2) * 32;
    const int mlane = lane & 15, half = lane >> 4;

    v8f ag[2][2], au[2][2];
#pragma unroll
    for (int i = 0; i < 2; ++i)
#pragma unroll
        for (int j = 0; j < 2; ++j) { ag[i][j] = (v8f)(0.0f); au[i][j] = (v8f)(0.0f); }

    for (int k0 = 0; k0 < K; k0 += 32) {
        if (k0 + 32 < K)
            __builtin_prefetch(A + (size_t)(m0 + mlane) * K + k0 + 32, 0, 0);
        v16bf a[2], bg[2], bu[2];
#pragma unroll
        for (int i = 0; i < 2; ++i) a[i] = load_a_frag(A, K, m0 + i * 16 + mlane, half, k0);
#pragma unroll
        for (int j = 0; j < 2; ++j) {
            bg[j] = load_b_frag(Wg, K, n0 + j * 16, lane, k0);
            bu[j] = load_b_frag(Wu, K, n0 + j * 16, lane, k0);
        }
#pragma unroll
        for (int i = 0; i < 2; ++i)
#pragma unroll
            for (int j = 0; j < 2; ++j) {
                ag[i][j] = __builtin_amdgcn_wmma_f32_16x16x32_bf16(
                    false, a[i], false, bg[j], (short)0, ag[i][j], false, false);
                au[i][j] = __builtin_amdgcn_wmma_f32_16x16x32_bf16(
                    false, a[i], false, bu[j], (short)0, au[i][j], false, false);
            }
    }
#pragma unroll
    for (int i = 0; i < 2; ++i)
#pragma unroll
        for (int j = 0; j < 2; ++j)
#pragma unroll
            for (int r = 0; r < 8; ++r) {
                int m = m0 + i * 16 + half * 8 + r;
                int n = n0 + j * 16 + mlane;
                float g = ag[i][j][r], u = au[i][j][r];
                // SiLU(g)*u via v_rcp_f32 (avoid IEEE div_scale/div_fmas chain)
                float s = g * fast_rcp(1.0f + fast_exp(-g)) * u;
                out[(size_t)m * N + n] = f2bf(s);
            }
}

// ---------------- RMSNorm (fp32 in, bf16*w out); one wave per 512-row ------
__global__ __launch_bounds__(256) void k_rmsnorm_bf16(const float* __restrict__ x,
                                                      const float* __restrict__ w,
                                                      unsigned short* __restrict__ out,
                                                      int R) {
    int row = blockIdx.x * 8 + (threadIdx.x >> 5);
    if (row >= R) return;
    int lane = threadIdx.x & 31;
    const float* xr = x + (size_t)row * HID + lane * 16;
    float buf[16]; float s = 0.0f;
#pragma unroll
    for (int t = 0; t < 16; ++t) { buf[t] = xr[t]; s += buf[t] * buf[t]; }
#pragma unroll
    for (int off = 16; off > 0; off >>= 1) s += __shfl_xor(s, off, 32);
    float inv = rsqrtf(s * (1.0f / HID) + 1e-6f);
    unsigned short* orow = out + (size_t)row * HID + lane * 16;
    const float* wr = w + lane * 16;
#pragma unroll
    for (int t = 0; t < 16; ++t) orow[t] = f2bf(buf[t] * inv * wr[t]);
}

// ---------------- elementwise helpers ----------------
__global__ void k_cast_bf16(const float* __restrict__ in, unsigned short* __restrict__ out,
                            long long n) {
    long long i = (long long)blockIdx.x * blockDim.x + threadIdx.x;
    long long stride = (long long)gridDim.x * blockDim.x;
    for (; i < n; i += stride) out[i] = f2bf(in[i]);
}
__global__ void k_fill(float* __restrict__ p, float v, long long n) {
    long long i = (long long)blockIdx.x * blockDim.x + threadIdx.x;
    long long stride = (long long)gridDim.x * blockDim.x;
    for (; i < n; i += stride) p[i] = v;
}

// ---------------- graph attention ----------------
__device__ __forceinline__ void atomicMaxF(float* addr, float val) {
    int* ia = (int*)addr;
    int old = __float_as_int(*addr);
    while (__int_as_float(old) < val) {
        int assumed = old;
        old = atomicCAS(ia, assumed, __float_as_int(val));
        if (old == assumed) break;
    }
}

// alpha[e,h,l] = rope(q[dst]) . rope(k[src]+ek) / 8 ; segment max into amax
__global__ __launch_bounds__(256) void k_edge_score(const float* __restrict__ q,
                                                    const float* __restrict__ k,
                                                    const float* __restrict__ ek,
                                                    const int* __restrict__ src,
                                                    const int* __restrict__ dst,
                                                    float* __restrict__ alpha,
                                                    float* __restrict__ amax) {
    int i = blockIdx.x * blockDim.x + threadIdx.x;
    if (i >= NEDGES * NH * QL) return;
    int e  = i / (NH * QL);
    int hh = (i / QL) % NH;
    int l  = i % QL;
    int sv = src[e], dv = dst[e];
    int kvh = hh >> 2;  // G = NH/KVH = 4

    const float* qp  = q  + ((size_t)dv * QL + l) * (NH * HD) + hh * HD;
    const float* kp  = k  + ((size_t)sv * QL + l) * KVW + kvh * HD;
    const float* ekp = ek + ((size_t)e  * QL + l) * KVW + kvh * HD;

    float acc = 0.0f;
    const float lf = (float)l;
#pragma unroll 8
    for (int d = 0; d < 32; ++d) {
        // inv_freq[d] = 10000^(-d/32) = exp2(-d * log2(10000)/32)
        float ang = lf * fast_exp2(-(float)d * 0.41524100776f);
        float c = __cosf(ang), s = __sinf(ang);
        float q1 = qp[d], q2 = qp[d + 32];
        float k1 = kp[d] + ekp[d], k2 = kp[d + 32] + ekp[d + 32];
        float qr1 = q1 * c - q2 * s, qr2 = q2 * c + q1 * s;
        float kr1 = k1 * c - k2 * s, kr2 = k2 * c + k1 * s;
        acc += qr1 * kr1 + qr2 * kr2;
    }
    acc *= 0.125f;  // 1/sqrt(64)
    alpha[i] = acc;
    atomicMaxF(&amax[((size_t)dv * NH + hh) * QL + l], acc);
}

// alpha <- exp(alpha - amax[dst]) ; den += alpha
__global__ __launch_bounds__(256) void k_edge_exp(const int* __restrict__ dst,
                                                  float* __restrict__ alpha,
                                                  const float* __restrict__ amax,
                                                  float* __restrict__ den) {
    int i = blockIdx.x * blockDim.x + threadIdx.x;
    if (i >= NEDGES * NH * QL) return;
    int e  = i / (NH * QL);
    int hh = (i / QL) % NH;
    int l  = i % QL;
    size_t ni = ((size_t)dst[e] * NH + hh) * QL + l;
    float ex = fast_exp(alpha[i] - amax[ni]);
    alpha[i] = ex;
    atomicAdd(&den[ni], ex);
}

// o[dst,l,h*64+d] += (v[src]+ev) * alpha/den  (one thread per (e,l,kv-dim))
__global__ __launch_bounds__(256) void k_edge_scatter(const int* __restrict__ src,
                                                      const int* __restrict__ dst,
                                                      const float* __restrict__ alpha,
                                                      const float* __restrict__ den,
                                                      const float* __restrict__ v,
                                                      const float* __restrict__ ev,
                                                      float* __restrict__ o) {
    long long i = (long long)blockIdx.x * blockDim.x + threadIdx.x;
    if (i >= (long long)NEDGES * QL * KVW) return;
    int e  = (int)(i / (QL * KVW));
    int l  = (int)((i / KVW) % QL);
    int dd = (int)(i % KVW);
    int kvh = dd >> 6, d = dd & 63;
    int sv = src[e], dv = dst[e];
    float vv = v[((size_t)sv * QL + l) * KVW + dd] + ev[((size_t)e * QL + l) * KVW + dd];
#pragma unroll
    for (int g = 0; g < 4; ++g) {
        int head = kvh * 4 + g;
        float wgt = alpha[((size_t)e * NH + head) * QL + l] *
                    fast_rcp(den[((size_t)dv * NH + head) * QL + l]);
        atomicAdd(&o[((size_t)dv * QL + l) * (NH * HD) + head * HD + d], vv * wgt);
    }
}

// ---------------- host orchestration ----------------
static inline void* carve(char*& p, size_t bytes) {
    void* r = (void*)p;
    p += (bytes + 255) & ~(size_t)255;
    return r;
}

extern "C" void kernel_launch(void* const* d_in, const int* in_sizes, int n_in,
                              void* d_out, int out_size, void* d_ws, size_t ws_size,
                              hipStream_t stream) {
    const float* hidden  = (const float*)d_in[0];
    const float* ehidden = (const float*)d_in[1];
    const int*   eidx    = (const int*)d_in[2];
    const float* ln1     = (const float*)d_in[3];
    const float* ln2     = (const float*)d_in[4];
    const float* wq      = (const float*)d_in[5];
    const float* wk      = (const float*)d_in[6];
    const float* wv      = (const float*)d_in[7];
    const float* wo      = (const float*)d_in[8];
    const float* wg      = (const float*)d_in[9];
    const float* wu      = (const float*)d_in[10];
    const float* wd      = (const float*)d_in[11];
    const int* src = eidx;
    const int* dst = eidx + NEDGES;

    char* p = (char*)d_ws;
    unsigned short* wq_bf  = (unsigned short*)carve(p, (size_t)HID * HID * 2);
    unsigned short* wk_bf  = (unsigned short*)carve(p, (size_t)KVW * HID * 2);
    unsigned short* wv_bf  = (unsigned short*)carve(p, (size_t)KVW * HID * 2);
    unsigned short* wo_bf  = (unsigned short*)carve(p, (size_t)HID * HID * 2);
    unsigned short* wg_bf  = (unsigned short*)carve(p, (size_t)INTERD * HID * 2);
    unsigned short* wu_bf  = (unsigned short*)carve(p, (size_t)INTERD * HID * 2);
    unsigned short* wd_bf  = (unsigned short*)carve(p, (size_t)HID * INTERD * 2);
    unsigned short* hn_bf  = (unsigned short*)carve(p, (size_t)NROWS * HID * 2);
    unsigned short* ehn_bf = (unsigned short*)carve(p, (size_t)EROWS * HID * 2);
    float* Q   = (float*)carve(p, (size_t)NROWS * HID * 4);
    float* Kp  = (float*)carve(p, (size_t)NROWS * KVW * 4);
    float* Vp  = (float*)carve(p, (size_t)NROWS * KVW * 4);
    float* EK  = (float*)carve(p, (size_t)EROWS * KVW * 4);
    float* EV  = (float*)carve(p, (size_t)EROWS * KVW * 4);
    float* ALPHA = (float*)carve(p, (size_t)NEDGES * NH * QL * 4);
    float* AMAX  = (float*)carve(p, (size_t)NNODES * NH * QL * 4);
    float* DEN   = (float*)carve(p, (size_t)NNODES * NH * QL * 4);
    float* O   = (float*)carve(p, (size_t)NROWS * HID * 4);
    unsigned short* o_bf   = (unsigned short*)carve(p, (size_t)NROWS * HID * 2);
    float* Hb  = (float*)carve(p, (size_t)NROWS * HID * 4);
    unsigned short* hn2_bf = (unsigned short*)carve(p, (size_t)NROWS * HID * 2);
    unsigned short* int_bf = (unsigned short*)carve(p, (size_t)NROWS * INTERD * 2);

    auto cast = [&](const float* in, unsigned short* out, long long n) {
        int blocks = (int)((n + 255) / 256); if (blocks > 4096) blocks = 4096;
        k_cast_bf16<<<blocks, 256, 0, stream>>>(in, out, n);
    };
    auto fill = [&](float* q_, float v_, long long n) {
        int blocks = (int)((n + 255) / 256); if (blocks > 4096) blocks = 4096;
        k_fill<<<blocks, 256, 0, stream>>>(q_, v_, n);
    };

    // 1) weights -> bf16 (weights are L2-resident thereafter: 7.6 MB total)
    cast(wq, wq_bf, (long long)HID * HID);
    cast(wk, wk_bf, (long long)KVW * HID);
    cast(wv, wv_bf, (long long)KVW * HID);
    cast(wo, wo_bf, (long long)HID * HID);
    cast(wg, wg_bf, (long long)INTERD * HID);
    cast(wu, wu_bf, (long long)INTERD * HID);
    cast(wd, wd_bf, (long long)HID * INTERD);

    // 2) RMSNorm -> bf16 activations
    k_rmsnorm_bf16<<<NROWS / 8, 256, 0, stream>>>(hidden, ln1, hn_bf, NROWS);
    k_rmsnorm_bf16<<<EROWS / 8, 256, 0, stream>>>(ehidden, ln1, ehn_bf, EROWS);

    // 3) projections (WMMA GEMMs)
    k_gemm<false><<<dim3(NROWS / 128, HID / 128), 256, 0, stream>>>(hn_bf, wq_bf, nullptr, Q,  NROWS, HID, HID);
    k_gemm<false><<<dim3(NROWS / 128, 1),         256, 0, stream>>>(hn_bf, wk_bf, nullptr, Kp, NROWS, KVW, HID);
    k_gemm<false><<<dim3(NROWS / 128, 1),         256, 0, stream>>>(hn_bf, wv_bf, nullptr, Vp, NROWS, KVW, HID);
    k_gemm<false><<<dim3(EROWS / 128, 1),         256, 0, stream>>>(ehn_bf, wk_bf, nullptr, EK, EROWS, KVW, HID);
    k_gemm<false><<<dim3(EROWS / 128, 1),         256, 0, stream>>>(ehn_bf, wv_bf, nullptr, EV, EROWS, KVW, HID);

    // 4) segment softmax over incoming edges
    fill(AMAX, -3.4e38f, (long long)NNODES * NH * QL);
    fill(DEN, 0.0f, (long long)NNODES * NH * QL);
    fill(O, 0.0f, (long long)NROWS * HID);
    {
        int tot = NEDGES * NH * QL;
        k_edge_score<<<(tot + 255) / 256, 256, 0, stream>>>(Q, Kp, EK, src, dst, ALPHA, AMAX);
        k_edge_exp<<<(tot + 255) / 256, 256, 0, stream>>>(dst, ALPHA, AMAX, DEN);
    }
    {
        long long tot = (long long)NEDGES * QL * KVW;
        k_edge_scatter<<<(int)((tot + 255) / 256), 256, 0, stream>>>(src, dst, ALPHA, DEN, Vp, EV, O);
    }

    // 5) output projection + residual -> Hb
    cast(O, o_bf, (long long)NROWS * HID);
    k_gemm<true><<<dim3(NROWS / 128, HID / 128), 256, 0, stream>>>(o_bf, wo_bf, hidden, Hb, NROWS, HID, HID);

    // 6) MLP: rmsnorm -> fused gate/up SiLU -> down + residual -> d_out
    k_rmsnorm_bf16<<<NROWS / 8, 256, 0, stream>>>(Hb, ln2, hn2_bf, NROWS);
    k_gemm_gateup<<<dim3(NROWS / 128, INTERD / 64), 256, 0, stream>>>(hn2_bf, wg_bf, wu_bf, int_bf, NROWS, INTERD, HID);
    k_gemm<true><<<dim3(NROWS / 128, HID / 128), 256, 0, stream>>>(int_bf, wd_bf, Hb, (float*)d_out, NROWS, HID, INTERD);
}
